// HMR_70729521430970
// MI455X (gfx1250) — matrix operand order repaired and checked by reading, not compiled, not run on hardware
//
#include <hip/hip_runtime.h>
#include <math.h>

#define B_ 64
#define V_ 2048
#define K_ 128
#define H_ 128
#define N_ (B_*V_)
#define E_ 1048576
#define CF_ 47
#define L_ 3
#define EPS_ 1e-5f

typedef __attribute__((ext_vector_type(16))) __bf16 v16bf;
typedef __attribute__((ext_vector_type(8)))  float  v8f;

__device__ __forceinline__ __bf16 to_bf16(float x) { return (__bf16)x; }
__device__ __forceinline__ float fast_rcp(float x) { return __builtin_amdgcn_rcpf(x); }
__device__ __forceinline__ float sigmoidf_(float x) { return fast_rcp(1.0f + __expf(-x)); }
__device__ __forceinline__ float siluf_(float x)    { return x * sigmoidf_(x); }
__device__ __forceinline__ float softplusf_(float x){ return (x > 20.0f) ? x : __logf(1.0f + __expf(x)); }

__device__ __forceinline__ v8f wmma_bf16(v16bf a, v16bf b, v8f c) {
    return __builtin_amdgcn_wmma_f32_16x16x32_bf16(false, a, false, b, (short)0, c, false, false);
}

__host__ __device__ constexpr int ilog2(int v) { int r = 0; while ((1 << r) < v) ++r; return r; }

// ---------------------------------------------------------------------------
// Generic bf16-WMMA GEMM: C[M x NT] = act(A[M x KT]) * W[KT x NT] (+bias)
// 8 waves / block, one 16-row tile per wave.
// W staged in LDS in *fragment order*: fragment g=(nt*NC+c), lane L owns 16
// contiguous bf16 at Wl[(g*32+L)*16], so each WMMA's B operand is one 32-byte
// vector LDS load. Lane layout per CDNA5 ISA: B K-index = 8*(L>>4) + (e&7) + 16*(e>>3),
// N = nt*16 + (L&15).
// ---------------------------------------------------------------------------
template<int KT, int KP, int NT, bool PRE, bool BIAS, bool STORE, bool STATS>
__global__ __launch_bounds__(256)
void gemm_bf16_kernel(const float* __restrict__ A, long long sAb, int lda,
                      const float* __restrict__ W, long long sWb,
                      const float* __restrict__ bias,
                      const float* __restrict__ ps, const float* __restrict__ pt,
                      float* __restrict__ C, long long sCb, int ldc,
                      float* __restrict__ stat_sum, float* __restrict__ stat_sq)
{
    constexpr int NC  = KP / 32;           // k-chunks (power of two)
    constexpr int LN  = ilog2(NT);
    __shared__ __align__(32) __bf16 Wl[KP * NT];
    __shared__ float s_sum[STATS ? NT : 1];
    __shared__ float s_sq [STATS ? NT : 1];
    __shared__ float s_ps [PRE ? KP : 1];
    __shared__ float s_pt [PRE ? KP : 1];

    const int tid  = threadIdx.x;
    const int wave = tid >> 5;
    const int lane = tid & 31;
    const int half = lane >> 4;
    const int col  = lane & 15;
    const int b    = blockIdx.y;

    // stage W swizzled: coalesced global reads, scattered LDS writes
    const float* Wg = W + (long long)b * sWb;
    for (int e = tid; e < KP * NT; e += 256) {
        const int k = e >> LN, n = e & (NT - 1);
        const float v = (k < KT) ? Wg[e] : 0.0f;
        const int c = k >> 5, kk = k & 31;
        const int dst = (((n >> 4) * NC + c) * 32 + ((kk >> 3) & 1) * 16 + (n & 15)) * 16 + (kk & 7) + ((kk >> 4) << 3);
        Wl[dst] = to_bf16(v);
    }
    if constexpr (STATS) {
        for (int e = tid; e < NT; e += 256) { s_sum[e] = 0.0f; s_sq[e] = 0.0f; }
    }
    if constexpr (PRE) {
        for (int e = tid; e < KP; e += 256) {
            s_ps[e] = (e < KT) ? ps[e] : 0.0f;
            s_pt[e] = (e < KT) ? pt[e] : 0.0f;
        }
    }
    __syncthreads();

    const int rowbase = (blockIdx.x * 8 + wave) * 16;
    const float* Arow = A + (long long)b * sAb + (long long)(rowbase + col) * lda;

    // A fragments: lane M = col; K = 8*half + {e, 16+e}
    v16bf af[NC];
#pragma unroll
    for (int c = 0; c < NC; ++c) {
        const int kb = c * 32 + 8 * half;
        float xs[16];
        if constexpr (KT == KP) {
            const float4 a0 = *(const float4*)(Arow + kb);
            const float4 a1 = *(const float4*)(Arow + kb + 4);
            const float4 a2 = *(const float4*)(Arow + kb + 16);
            const float4 a3 = *(const float4*)(Arow + kb + 20);
            xs[0]=a0.x; xs[1]=a0.y; xs[2]=a0.z; xs[3]=a0.w;
            xs[4]=a1.x; xs[5]=a1.y; xs[6]=a1.z; xs[7]=a1.w;
            xs[8]=a2.x; xs[9]=a2.y; xs[10]=a2.z; xs[11]=a2.w;
            xs[12]=a3.x; xs[13]=a3.y; xs[14]=a3.z; xs[15]=a3.w;
        } else {
#pragma unroll
            for (int e = 0; e < 8; ++e) {
                const int k1 = kb + e, k2 = kb + 16 + e;
                xs[e]     = (k1 < KT) ? Arow[k1] : 0.0f;
                xs[e + 8] = (k2 < KT) ? Arow[k2] : 0.0f;
            }
        }
#pragma unroll
        for (int e = 0; e < 16; ++e) {
            float x = xs[e];
            if constexpr (PRE) {
                const int k = kb + ((e < 8) ? e : (8 + e));
                x = siluf_(s_ps[k] * x + s_pt[k]);
            }
            af[c][e] = to_bf16(x);
        }
    }

    const v16bf* __restrict__ Wf = (const v16bf*)Wl;
    for (int nt = 0; nt < NT / 16; ++nt) {
        const int n0 = nt * 16;
        v8f acc = {};
#pragma unroll
        for (int c = 0; c < NC; ++c) {
            const v16bf bf = Wf[(nt * NC + c) * 32 + lane];
            acc = wmma_bf16(af[c], bf, acc);
        }
        const int n = n0 + col;
        float bia = 0.0f;
        if constexpr (BIAS) bia = bias[n];
        float lsum = 0.0f, lsq = 0.0f;
#pragma unroll
        for (int j = 0; j < 8; ++j) {
            const float v = acc[j] + bia;
            if constexpr (STORE)
                C[(long long)b * sCb + (long long)(rowbase + j + 8 * half) * ldc + n] = v;
            if constexpr (STATS) { lsum += v; lsq += v * v; }
        }
        if constexpr (STATS) {
            atomicAdd(&s_sum[n], lsum);
            atomicAdd(&s_sq[n],  lsq);
        }
    }
    if constexpr (STATS) {
        __syncthreads();
        for (int e = tid; e < NT; e += 256) {
            atomicAdd(&stat_sum[e], s_sum[e]);
            atomicAdd(&stat_sq[e],  s_sq[e]);
        }
    }
}

// ---------------------------------------------------------------------------
// GEMM2 apply: y=silu(bn1(h1)) @ w2 + b2 -> bn2 -> msg=sig(F)*softplus(Co)
// -> atomic scatter-add into h[nbr[row]]. Pairs column tiles (c, c+128).
// ---------------------------------------------------------------------------
__global__ __launch_bounds__(256)
void gemm2_apply_kernel(const float* __restrict__ h1,
                        const float* __restrict__ w2,
                        const float* __restrict__ b2,
                        const float* __restrict__ aS, const float* __restrict__ aT,
                        const float* __restrict__ bS, const float* __restrict__ bT,
                        const int*   __restrict__ nbr,
                        float* __restrict__ hout)
{
    __shared__ __align__(32) __bf16 Wl[128 * 256];
    __shared__ float s_ps[128], s_pt[128];
    const int tid = threadIdx.x;
    for (int e = tid; e < 128 * 256; e += 256) {
        const int k = e >> 8, n = e & 255;
        const int c = k >> 5, kk = k & 31;
        const int dst = (((n >> 4) * 4 + c) * 32 + ((kk >> 3) & 1) * 16 + (n & 15)) * 16 + (kk & 7) + ((kk >> 4) << 3);
        Wl[dst] = to_bf16(w2[e]);
    }
    for (int e = tid; e < 128; e += 256) { s_ps[e] = aS[e]; s_pt[e] = aT[e]; }
    __syncthreads();

    const int wave = tid >> 5, lane = tid & 31, half = lane >> 4, col = lane & 15;
    const int rowbase = (blockIdx.x * 8 + wave) * 16;
    const float* Arow = h1 + (long long)(rowbase + col) * 128;

    v16bf af[4];
#pragma unroll
    for (int c = 0; c < 4; ++c) {
        const int kb = c * 32 + 8 * half;
        const float4 a0 = *(const float4*)(Arow + kb);
        const float4 a1 = *(const float4*)(Arow + kb + 4);
        const float4 a2 = *(const float4*)(Arow + kb + 16);
        const float4 a3 = *(const float4*)(Arow + kb + 20);
        float xs[16] = {a0.x,a0.y,a0.z,a0.w, a1.x,a1.y,a1.z,a1.w,
                        a2.x,a2.y,a2.z,a2.w, a3.x,a3.y,a3.z,a3.w};
#pragma unroll
        for (int e = 0; e < 16; ++e) {
            const int k = kb + ((e < 8) ? e : (8 + e));
            af[c][e] = to_bf16(siluf_(s_ps[k] * xs[e] + s_pt[k]));
        }
    }
    int seg[8];
#pragma unroll
    for (int j = 0; j < 8; ++j) seg[j] = nbr[rowbase + j + 8 * half];

    const v16bf* __restrict__ Wf = (const v16bf*)Wl;
    for (int jt = 0; jt < 8; ++jt) {
        v8f accF = {}; v8f accC = {};
#pragma unroll
        for (int c = 0; c < 4; ++c) {
            const v16bf bF = Wf[(jt * 4 + c) * 32 + lane];
            const v16bf bC = Wf[((jt + 8) * 4 + c) * 32 + lane];
            accF = wmma_bf16(af[c], bF, accF);
            accC = wmma_bf16(af[c], bC, accC);
        }
        const int nf = jt * 16 + col, nc2 = 128 + jt * 16 + col;
        const float sF = bS[nf],  tF = bT[nf],  bf2 = b2[nf];
        const float sC = bS[nc2], tC = bT[nc2], bc2 = b2[nc2];
#pragma unroll
        for (int j = 0; j < 8; ++j) {
            const float f   = sF * (accF[j] + bf2) + tF;
            const float co  = sC * (accC[j] + bc2) + tC;
            const float msg = sigmoidf_(f) * softplusf_(co);
            atomicAdd(&hout[(long long)seg[j] * 128 + nf], msg);
        }
    }
}

// ---------------------------------------------------------------------------
// Spectral projection: hspec[b,k,n] = exp(-eig[b,k]*time[n]) *
//                                     sum_v evinv[b,v,k] * h[b,v,n]
// Both LDS chunks stored pre-swizzled in WMMA fragment order so the inner
// reduction loop is: 2 vector LDS loads + 1 WMMA per 32-row chunk.
// ---------------------------------------------------------------------------
__global__ __launch_bounds__(256)
void spectral_proj_kernel(const float* __restrict__ h,
                          const float* __restrict__ eigs,
                          const float* __restrict__ ptime,
                          float* __restrict__ hspec)
{
    __shared__ __align__(32) __bf16 evl[32 * 16];    // A fragments (16 k-rows x 32 v)
    __shared__ __align__(32) __bf16 hbl[32 * 128];   // B fragments (32 v x 128 n)
    const int tid = threadIdx.x;
    const int wave = tid >> 5, lane = tid & 31, half = lane >> 4, col = lane & 15;
    const int b = blockIdx.y, kt = blockIdx.x;
    const float* eb = eigs + (long long)b * 4097 * 128;

    v8f acc = {};
    for (int vc = 0; vc < 64; ++vc) {
        const int vbase = vc * 32;
        // B: lane needs v = 8*(L>>4)+(e&7)+16*(e>>3), n = nt*16 + (L&15); frag g = nt
        for (int e = tid; e < 4096; e += 256) {
            const int r = e >> 7, n = e & 127;
            const float v = h[((long long)b * V_ + vbase + r) * 128 + n];
            const int dst = ((n >> 4) * 32 + ((r >> 3) & 1) * 16 + (n & 15)) * 16 + (r & 7) + ((r >> 4) << 3);
            hbl[dst] = to_bf16(v);
        }
        // A: lane L = half*16 + m; v-index: half=(v>>3)&1, e2=(v<16)?(v&7):((v&7)+8)
        for (int e = tid; e < 512; e += 256) {
            const int r = e >> 4, m = e & 15;
            const float v = eb[(long long)(1 + V_ + vbase + r) * 128 + kt * 16 + m];
            const int ha = (r >> 3) & 1;
            const int e2 = (r < 16) ? (r & 7) : ((r & 7) + 8);
            evl[(ha * 16 + m) * 16 + e2] = to_bf16(v);
        }
        // prefetch next chunk of h (256 threads cover 32 rows x 8 cachelines)
        if (vc + 1 < 64) {
            const float* nxt = h + ((long long)b * V_ + vbase + 32 + (tid >> 3)) * 128 + (tid & 7) * 16;
            __builtin_prefetch(nxt, 0, 1);
        }
        __syncthreads();
        const v16bf a  = *(const v16bf*)(evl + lane * 16);
        const v16bf bf = *(const v16bf*)(hbl + (wave * 32 + lane) * 16);
        acc = wmma_bf16(a, bf, acc);
        __syncthreads();
    }

    const int n = wave * 16 + col;
    const float tm = fmaxf(ptime[n], 1e-6f);
#pragma unroll
    for (int j = 0; j < 8; ++j) {
        const int m = kt * 16 + j + 8 * half;
        hspec[((long long)b * 128 + m) * 128 + n] = acc[j] * __expf(-eb[m] * tm);
    }
}

// --------------------------- small utility kernels --------------------------
__global__ void zero_kernel(float* p, long long n) {
    long long i = (long long)blockIdx.x * blockDim.x + threadIdx.x;
    const long long st = (long long)gridDim.x * blockDim.x;
    for (; i < n; i += st) p[i] = 0.0f;
}

__global__ void finalize_kernel(const float* __restrict__ sum, const float* __restrict__ sq,
                                const float* __restrict__ g, const float* __restrict__ be,
                                float invR, int nc,
                                float* __restrict__ af_s, float* __restrict__ af_t)
{
    const int n = blockIdx.x * blockDim.x + threadIdx.x;
    if (n < nc) {
        const float m  = sum[n] * invR;
        const float v  = sq[n] * invR - m * m;
        const float s  = g[n] * rsqrtf(v + EPS_);
        af_s[n] = s;
        af_t[n] = be[n] - m * s;
    }
}

__global__ __launch_bounds__(256)
void h_to_zcat_kernel(const float* __restrict__ h, float* __restrict__ zc) {
    const long long i = (long long)blockIdx.x * 256 + threadIdx.x;
    const int row = (int)(i >> 7), c = (int)(i & 127);
    zc[(long long)row * 256 + c] = h[i];
}

__global__ __launch_bounds__(256)
void resid_kernel(float* __restrict__ h, const float* __restrict__ t2,
                  const float* __restrict__ s, const float* __restrict__ t,
                  float* __restrict__ zc)
{
    const long long i = (long long)blockIdx.x * 256 + threadIdx.x;
    const int row = (int)(i >> 7), c = (int)(i & 127);
    const float v = h[i] + (s[c] * t2[i] + t[c]);
    h[i] = v;
    zc[(long long)row * 256 + c] = v;
}

__global__ __launch_bounds__(256)
void final_reduce_kernel(const float* __restrict__ t, const float* __restrict__ s,
                         const float* __restrict__ tt, const float* __restrict__ wo2,
                         float* __restrict__ oacc)
{
    const int r = blockIdx.x * 256 + threadIdx.x;
    const float* row = t + (long long)r * H_;
    float acc = 0.0f;
#pragma unroll 8
    for (int n = 0; n < H_; ++n) {
        const float x = s[n] * row[n] + tt[n];
        acc += siluf_(x) * wo2[n];
    }
    atomicAdd(&oacc[r >> 11], acc);
}

__global__ void writeout_kernel(const float* __restrict__ oacc,
                                const float* __restrict__ bo2, float* __restrict__ out) {
    const int b = threadIdx.x;
    if (b < B_) out[b] = oacc[b] * (1.0f / V_) + bo2[0];
}

// ---------------------------------------------------------------------------
extern "C" void kernel_launch(void* const* d_in, const int* in_sizes, int n_in,
                              void* d_out, int out_size, void* d_ws, size_t ws_size,
                              hipStream_t stream)
{
    (void)in_sizes; (void)n_in; (void)out_size; (void)ws_size;
    const float* chem = (const float*)d_in[0];
    const int*   nbr  = (const int*)  d_in[1];
    const float* eigs = (const float*)d_in[2];
    const float* w1   = (const float*)d_in[3];
    const float* b1   = (const float*)d_in[4];
    const float* g1   = (const float*)d_in[5];
    const float* be1  = (const float*)d_in[6];
    const float* w2   = (const float*)d_in[7];
    const float* b2   = (const float*)d_in[8];
    const float* g2   = (const float*)d_in[9];
    const float* be2  = (const float*)d_in[10];
    const float* ptm  = (const float*)d_in[11];
    const float* pw1  = (const float*)d_in[12];
    const float* pb1  = (const float*)d_in[13];
    const float* pg1  = (const float*)d_in[14];
    const float* pbe1 = (const float*)d_in[15];
    const float* pw2  = (const float*)d_in[16];
    const float* pb2  = (const float*)d_in[17];
    const float* pg2  = (const float*)d_in[18];
    const float* pbe2 = (const float*)d_in[19];
    const float* wo1  = (const float*)d_in[20];
    const float* bo1  = (const float*)d_in[21];
    const float* go1  = (const float*)d_in[22];
    const float* beo1 = (const float*)d_in[23];
    const float* wo2  = (const float*)d_in[24];
    const float* bo2  = (const float*)d_in[25];
    float* out = (float*)d_out;

    float* ws  = (float*)d_ws;
    size_t off = 0;
    float* h1  = ws + off; off += (size_t)E_ * H_;
    float* h   = ws + off; off += (size_t)N_ * H_;
    float* zc  = ws + off; off += (size_t)N_ * 2 * H_;
    float* t1  = ws + off; off += (size_t)N_ * H_;
    float* t2  = ws + off; off += (size_t)N_ * H_;
    float* hs  = ws + off; off += (size_t)B_ * K_ * H_;
    float* stA = ws + off; off += 256;
    float* afA = ws + off; off += 256;
    float* stB = ws + off; off += 512;
    float* afB = ws + off; off += 512;
    float* stP = ws + off; off += 256;
    float* afP = ws + off; off += 256;
    float* stQ = ws + off; off += 256;
    float* afQ = ws + off; off += 256;
    float* stO = ws + off; off += 256;
    float* afO = ws + off; off += 256;
    float* oacc = ws + off; off += 64;

    // --- init ---
    zero_kernel<<<2048, 256, 0, stream>>>(h, (long long)N_ * H_);
    zero_kernel<<<1, 256, 0, stream>>>(stA, 256);
    zero_kernel<<<2, 256, 0, stream>>>(stB, 512);
    zero_kernel<<<1, 64, 0, stream>>>(oacc, 64);

    // --- chem MLP ---
    gemm_bf16_kernel<CF_, 64, H_, false, true, true, true>
        <<<dim3(E_ / 128, 1), 256, 0, stream>>>(
            chem, 0, CF_, w1, 0, b1, nullptr, nullptr, h1, 0, H_, stA, stA + H_);
    finalize_kernel<<<1, 256, 0, stream>>>(stA, stA + H_, g1, be1, 1.0f / E_, H_, afA, afA + H_);

    gemm_bf16_kernel<H_, H_, 2 * H_, true, true, false, true>
        <<<dim3(E_ / 128, 1), 256, 0, stream>>>(
            h1, 0, H_, w2, 0, b2, afA, afA + H_, h1, 0, 2 * H_, stB, stB + 2 * H_);
    finalize_kernel<<<1, 256, 0, stream>>>(stB, stB + 2 * H_, g2, be2, 1.0f / E_, 2 * H_, afB, afB + 2 * H_);

    gemm2_apply_kernel<<<E_ / 128, 256, 0, stream>>>(
        h1, w2, b2, afA, afA + H_, afB, afB + 2 * H_, nbr, h);

    h_to_zcat_kernel<<<(N_ * H_) / 256, 256, 0, stream>>>(h, zc);

    // --- propagation layers ---
    for (int l = 0; l < L_; ++l) {
        spectral_proj_kernel<<<dim3(8, B_), 256, 0, stream>>>(h, eigs, ptm + l * H_, hs);

        // h_prop = evecs @ (coeff*hspec) -> zcat[:, 128:256]
        gemm_bf16_kernel<K_, K_, H_, false, false, true, false>
            <<<dim3(V_ / 128, B_), 256, 0, stream>>>(
                eigs + H_, (long long)4097 * H_, K_,
                hs, (long long)K_ * H_,
                nullptr, nullptr, nullptr,
                zc + H_, (long long)V_ * 2 * H_, 2 * H_, nullptr, nullptr);

        zero_kernel<<<1, 256, 0, stream>>>(stP, 256);
        gemm_bf16_kernel<2 * H_, 2 * H_, H_, false, true, true, true>
            <<<dim3(N_ / 128, 1), 256, 0, stream>>>(
                zc, 0, 2 * H_, pw1 + (long long)l * 2 * H_ * H_, 0, pb1 + l * H_,
                nullptr, nullptr, t1, 0, H_, stP, stP + H_);
        finalize_kernel<<<1, 256, 0, stream>>>(stP, stP + H_, pg1 + l * H_, pbe1 + l * H_,
                                               1.0f / N_, H_, afP, afP + H_);

        zero_kernel<<<1, 256, 0, stream>>>(stQ, 256);
        gemm_bf16_kernel<H_, H_, H_, true, true, true, true>
            <<<dim3(N_ / 128, 1), 256, 0, stream>>>(
                t1, 0, H_, pw2 + (long long)l * H_ * H_, 0, pb2 + l * H_,
                afP, afP + H_, t2, 0, H_, stQ, stQ + H_);
        finalize_kernel<<<1, 256, 0, stream>>>(stQ, stQ + H_, pg2 + l * H_, pbe2 + l * H_,
                                               1.0f / N_, H_, afQ, afQ + H_);

        resid_kernel<<<(N_ * H_) / 256, 256, 0, stream>>>(h, t2, afQ, afQ + H_, zc);
    }

    // --- output head ---
    zero_kernel<<<1, 256, 0, stream>>>(stO, 256);
    gemm_bf16_kernel<H_, H_, H_, false, true, true, true>
        <<<dim3(N_ / 128, 1), 256, 0, stream>>>(
            h, 0, H_, wo1, 0, bo1, nullptr, nullptr, t1, 0, H_, stO, stO + H_);
    finalize_kernel<<<1, 256, 0, stream>>>(stO, stO + H_, go1, beo1, 1.0f / N_, H_, afO, afO + H_);

    final_reduce_kernel<<<N_ / 256, 256, 0, stream>>>(t1, afO, afO + H_, wo2, oacc);
    writeout_kernel<<<1, 64, 0, stream>>>(oacc, bo2, out);
}
